// LSTMLayer_79164837200638
// MI455X (gfx1250) — matrix operand order repaired and checked by reading, not compiled, not run on hardware
//
#include <hip/hip_runtime.h>

// ---------------------------------------------------------------------------
// LSTM layer for MI455X (gfx1250, wave32, WMMA + TDM).
//
// 550 GFLOP total across 512 dependent steps -> sync/latency bound; W (16 MB
// bf16) + A (512 KB) live in the 192 MB L2. Strategy:
//   * prologue: transpose+convert gate weights into bf16 Wt[N=4096][K=2048]
//     (K contiguous -> contiguous 16B B-fragment loads)
//   * persistent kernel, 32 WGs x 256 threads (8 waves), one grid barrier per
//     timestep; double-buffered bf16 activation matrix A[64][2048] in global
//     (x slab | H slab): GEMM reads buf p while the update writes buf 1-p.
//   * per step, TENSOR_LOAD_TO_LDS DMAs the 256 KB A matrix into LDS; the
//     k-loop then feeds v_wmma_f32_16x16x32_bf16 from ds_load_b128 (A) +
//     global_load_b128 (W, L2-resident), fp32 accumulation.
//   * 32 KB gate buffer aliases the A LDS region (A reads complete first).
// ---------------------------------------------------------------------------

typedef __attribute__((ext_vector_type(16))) __bf16 v16bf;
typedef __attribute__((ext_vector_type(8)))  __bf16 v8bf;
typedef __attribute__((ext_vector_type(8)))  float  v8f;
typedef __attribute__((ext_vector_type(4)))  unsigned int u32x4;
typedef __attribute__((ext_vector_type(8)))  int i32x8;
typedef __attribute__((ext_vector_type(4)))  int i32x4;

#define T_STEPS 512
#define BATCH   64
#define VDIM    1024
#define HDIM    1024
#define KDIM    (VDIM + HDIM)   // 2048
#define NDIM    (4 * HDIM)      // 4096
#define NWG     32              // H columns split: 32 per WG
#define COLS    32              // H columns per WG
#define LDSW    128             // 4 gates * 32 cols
#define NTHREADS 256

// ---- workspace layout (bytes) ----
#define WS_WT   0u                          // bf16 Wt[4096][2048] = 16 MB
#define WS_A0   (16777216u)                 // bf16 A0[64][2048]  = 256 KB
#define WS_A1   (WS_A0 + 262144u)           // bf16 A1[64][2048]  = 256 KB
#define WS_C    (WS_A1 + 262144u)           // f32  C[64][1024]   = 256 KB
#define WS_SYNC (WS_C + 262144u)            // u32 barrier counter

__device__ __forceinline__ float sigmoidf_(float x) {
  return 1.0f / (1.0f + __expf(-x));
}

__device__ __forceinline__ v16bf combine16(v8bf lo, v8bf hi) {
  union { v16bf v; v8bf h[2]; } u;
  u.h[0] = lo; u.h[1] = hi;
  return u.v;
}

// Monotonic-counter grid barrier (no reset race); counter zeroed in prologue.
__device__ __forceinline__ void grid_barrier(unsigned* cnt, unsigned target) {
  __syncthreads();
  if (threadIdx.x == 0) {
    __threadfence();                 // release our stores to device scope
    atomicAdd(cnt, 1u);
    volatile unsigned* v = (volatile unsigned*)cnt;
    while (*v < target) { __builtin_amdgcn_s_sleep(1); }
  }
  __syncthreads();
  __threadfence();                   // acquire: invalidate stale near caches
}

// TDM: DMA a 2D bf16 tile [rows x width] (row stride == width, elements) from
// global memory into LDS. Descriptor bitfields per CDNA5 ISA ch.8 (D# groups).
__device__ __forceinline__ void tdm_load_2d_bf16(unsigned lds_addr,
                                                 const void* gptr,
                                                 unsigned width_elems,
                                                 unsigned rows) {
  unsigned long long ga = (unsigned long long)(__UINTPTR_TYPE__)gptr;
  u32x4 g0;
  g0[0] = 1u;                                             // count=1, user mode
  g0[1] = lds_addr;                                       // lds_addr [63:32]
  g0[2] = (unsigned)(ga & 0xFFFFFFFFull);                 // global_addr [95:64]
  g0[3] = (unsigned)((ga >> 32) & 0x01FFFFFFull)          // global_addr [120:96]
          | (2u << 30);                                   // type=2 ("image")
  i32x8 g1;
  g1[0] = (int)(1u << 16);                                // data_size=1 -> 2 bytes
  g1[1] = (int)((width_elems & 0xFFFFu) << 16);           // tensor_dim0[15:0]
  g1[2] = (int)(((width_elems >> 16) & 0xFFFFu)           // tensor_dim0[31:16]
          | ((rows & 0xFFFFu) << 16));                    // tensor_dim1[15:0]
  g1[3] = (int)(((rows >> 16) & 0xFFFFu)                  // tensor_dim1[31:16]
          | ((width_elems & 0xFFFFu) << 16));             // tile_dim0
  g1[4] = (int)(rows & 0xFFFFu);                          // tile_dim1 (tile_dim2=0)
  g1[5] = (int)width_elems;                               // tensor_dim0_stride[31:0]
  g1[6] = 0;                                              // stride[47:32], dim1_stride
  g1[7] = 0;
  i32x4 gz = {0, 0, 0, 0};
#if __clang_major__ >= 23
  i32x8 gz8 = {0, 0, 0, 0, 0, 0, 0, 0};
  __builtin_amdgcn_tensor_load_to_lds(g0, g1, gz, gz, gz8, 0);
#else
  __builtin_amdgcn_tensor_load_to_lds(g0, g1, gz, gz, 0);
#endif
}

// --- prologue 1: Wt[g*HDIM + j][k] = (bf16) Wg[k][j] ------------------------
__global__ void prep_weights(const float* __restrict__ Wf,
                             const float* __restrict__ Wi,
                             const float* __restrict__ Wc,
                             const float* __restrict__ Wo,
                             __bf16* __restrict__ Wt) {
  size_t idx = (size_t)blockIdx.x * blockDim.x + threadIdx.x;
  if (idx >= (size_t)NDIM * KDIM) return;
  int g = (int)(idx / ((size_t)HDIM * KDIM));
  size_t rem = idx - (size_t)g * HDIM * KDIM;
  int j = (int)(rem / KDIM);
  int k = (int)(rem % KDIM);
  const float* W = (g == 0) ? Wf : (g == 1) ? Wi : (g == 2) ? Wc : Wo;
  Wt[idx] = (__bf16)W[(size_t)k * HDIM + j];
}

// --- prologue 2: A0 = [bf16(x_0) | bf16(H0)], C = C0, sync = 0 --------------
__global__ void prep_state(const float* __restrict__ x0,
                           const float* __restrict__ H0,
                           const float* __restrict__ C0,
                           __bf16* __restrict__ A0,
                           float* __restrict__ Cbuf,
                           unsigned* __restrict__ cnt) {
  int i = blockIdx.x * blockDim.x + threadIdx.x;
  if (i >= BATCH * HDIM) return;
  int b = i >> 10;
  int j = i & 1023;
  A0[(size_t)b * KDIM + j]        = (__bf16)x0[i];   // x slab (V == Hd == 1024)
  A0[(size_t)b * KDIM + VDIM + j] = (__bf16)H0[i];   // H slab
  Cbuf[i] = C0[i];
  if (i == 0) *cnt = 0u;
}

// --- persistent recurrent kernel --------------------------------------------
__global__ __launch_bounds__(NTHREADS, 1)
void lstm_persistent(const float* __restrict__ inputs,
                     const __bf16* __restrict__ Wt,
                     __bf16* __restrict__ Abuf0,
                     __bf16* __restrict__ Abuf1,
                     float* __restrict__ Cbuf,
                     const float* __restrict__ bfv,
                     const float* __restrict__ biv,
                     const float* __restrict__ bcv,
                     const float* __restrict__ bov,
                     float* __restrict__ out,
                     unsigned* __restrict__ cnt) {
  // 256 KB: A matrix staged by TDM; first 32 KB reused for gate pre-acts
  // after all A reads complete (extra __syncthreads separates the phases).
  __shared__ __align__(16) unsigned char smem[BATCH * KDIM * 2];
  __bf16* aLDS = (__bf16*)smem;
  float*  gLDS = (float*)smem;

  const int wg   = blockIdx.x;              // owns H cols [wg*32, wg*32+32)
  const int tid  = threadIdx.x;
  const int wave = tid >> 5;                // 8 waves
  const int lane = tid & 31;
  const int lm   = lane & 15;
  const int hi   = lane >> 4;
  const int rt   = wave & 3;                // row tile (16 rows of B=64)
  const int grp  = wave >> 2;               // 0/1 -> n-tiles grp*4 .. grp*4+3

  // B-fragment row base pointers for this wave's 4 output n-tiles.
  // B 32x16 bf16 layout: lanes 0-15 hold K {k0..k0+7, k0+16..k0+23} of col n,
  // lanes 16-31 hold K {k0+8..k0+15, k0+24..k0+31}  ->  base + hi*8, two 16-byte loads.
  const __bf16* browB[4];
#pragma unroll
  for (int i = 0; i < 4; ++i) {
    int nt = grp * 4 + i;                   // 0..7
    int g  = nt >> 1;                       // gate 0..3
    int cs = nt & 1;                        // col sub-tile
    int n0 = g * HDIM + wg * COLS + cs * 16;
    browB[i] = Wt + (size_t)(n0 + lm) * KDIM + hi * 8;
  }

  // A 16x32 bf16 layout: lanes 0-15 row m, K {k0..k0+7, k0+16..k0+23};
  // lanes 16-31 row m, K {k0+8..k0+15, k0+24..k0+31} -> base + hi*8.
  const __bf16* arow = aLDS + (size_t)(rt * 16 + lm) * KDIM + hi * 8;
  const unsigned aLDSaddr = (unsigned)(__UINTPTR_TYPE__)aLDS;

  unsigned barrier_target = 0;

  for (int t = 0; t < T_STEPS; ++t) {
    const __bf16* Acur = (t & 1) ? Abuf1 : Abuf0;
    __bf16*       Anxt = (t & 1) ? Abuf0 : Abuf1;

    // ---------------- TDM: stage A(64x2048 bf16) into LDS -------------------
    if (wave == 0) {
      tdm_load_2d_bf16(aLDSaddr, Acur, KDIM, BATCH);
      __builtin_amdgcn_s_wait_tensorcnt(0);
    }
    __syncthreads();

    // ---------------- GEMM: g = A(64xK) * Wt^T, this WG's 4x32-wide slab ----
    v8f acc[4];
#pragma unroll
    for (int i = 0; i < 4; ++i) { v8f z = {}; acc[i] = z; }

    for (int k0 = 0; k0 < KDIM; k0 += 32) {
      v8bf alo = *(const v8bf*)(arow + k0);           // ds_load_b128
      v8bf ahi = *(const v8bf*)(arow + k0 + 16);
      v16bf afrag = combine16(alo, ahi);
#pragma unroll
      for (int i = 0; i < 4; ++i) {
        v8bf blo = *(const v8bf*)(browB[i] + k0);     // global_load_b128 (L2)
        v8bf bhi = *(const v8bf*)(browB[i] + k0 + 16);
        v16bf bfrag = combine16(blo, bhi);
        acc[i] = __builtin_amdgcn_wmma_f32_16x16x32_bf16(
            false, afrag, false, bfrag, (short)0, acc[i], false, false);
      }
    }

    // All waves done reading aLDS before the gate buffer overwrites it.
    __syncthreads();

    // C/D layout: VGPR r, lane l -> M = r + 8*(l>=16), N = l&15
#pragma unroll
    for (int i = 0; i < 4; ++i) {
      int nt = grp * 4 + i;
#pragma unroll
      for (int r = 0; r < 8; ++r) {
        int M = rt * 16 + hi * 8 + r;
        int N = nt * 16 + lm;
        gLDS[M * LDSW + N] = acc[i][r];
      }
    }
    __syncthreads();

    // ---------------- gates + recurrence (WG-private columns) ---------------
    const bool last = (t == T_STEPS - 1);
#pragma unroll 4
    for (int i = 0; i < 8; ++i) {
      int e  = i * NTHREADS + tid;          // 0..2047 = 64 rows x 32 cols
      int b  = e >> 5;
      int jl = e & 31;
      int jg = wg * COLS + jl;
      float fpre = gLDS[b * LDSW +  0 + jl] + bfv[jg];
      float ipre = gLDS[b * LDSW + 32 + jl] + biv[jg];
      float cpre = gLDS[b * LDSW + 64 + jl] + bcv[jg];
      float opre = gLDS[b * LDSW + 96 + jl] + bov[jg];
      float F  = sigmoidf_(fpre);
      float I  = sigmoidf_(ipre);
      float Cc = tanhf(cpre);
      float O  = sigmoidf_(opre);
      int ci = b * HDIM + jg;
      float Cn = Cbuf[ci] * F + I * Cc;
      float Hn = tanhf(Cn) * O;
      Cbuf[ci] = Cn;
      out[(size_t)t * (BATCH * HDIM) + ci] = Hn;          // states[t]
      Anxt[(size_t)b * KDIM + VDIM + jg] = (__bf16)Hn;    // H slab of next A
      if (last) {
        size_t S = (size_t)T_STEPS * BATCH * HDIM;
        out[S + ci] = Hn;                                 // Hf
        out[S + (size_t)BATCH * HDIM + ci] = Cn;          // Cf
      }
    }

    // ---------------- convert our 1/32 share of x(t+1) into next A ----------
    if (t + 1 < T_STEPS) {
      const float* xn = inputs + (size_t)(t + 1) * (BATCH * VDIM);
#pragma unroll 4
      for (int i = 0; i < 8; ++i) {
        int q   = wg * (BATCH * VDIM / NWG) + i * NTHREADS + tid;  // 0..65535
        int b   = q >> 10;
        int col = q & 1023;
        Anxt[(size_t)b * KDIM + col] = (__bf16)xn[b * VDIM + col];
      }
    }

    barrier_target += NWG;
    grid_barrier(cnt, barrier_target);
  }
}

// ---------------------------------------------------------------------------
extern "C" void kernel_launch(void* const* d_in, const int* in_sizes, int n_in,
                              void* d_out, int out_size, void* d_ws, size_t ws_size,
                              hipStream_t stream) {
  const float* inputs = (const float*)d_in[0];
  const float* H0     = (const float*)d_in[1];
  const float* C0     = (const float*)d_in[2];
  const float* Wf     = (const float*)d_in[3];
  const float* bf_    = (const float*)d_in[4];
  const float* Wi     = (const float*)d_in[5];
  const float* bi_    = (const float*)d_in[6];
  const float* Wc     = (const float*)d_in[7];
  const float* bc_    = (const float*)d_in[8];
  const float* Wo     = (const float*)d_in[9];
  const float* bo_    = (const float*)d_in[10];

  char* ws = (char*)d_ws;
  __bf16*   Wt  = (__bf16*)(ws + WS_WT);
  __bf16*   A0  = (__bf16*)(ws + WS_A0);
  __bf16*   A1  = (__bf16*)(ws + WS_A1);
  float*    Cb  = (float*)(ws + WS_C);
  unsigned* cnt = (unsigned*)(ws + WS_SYNC);

  // prologue: bf16 weight transpose (one-time per launch) + state init
  prep_weights<<<(NDIM * KDIM) / 256, 256, 0, stream>>>(Wf, Wi, Wc, Wo, Wt);
  prep_state<<<(BATCH * HDIM) / 256, 256, 0, stream>>>(inputs, H0, C0, A0, Cb, cnt);

  // persistent recurrent kernel: 32 WGs x 256 threads, 1 grid barrier / step
  lstm_persistent<<<NWG, NTHREADS, 0, stream>>>(
      inputs, Wt, A0, A1, Cb, bf_, bi_, bc_, bo_, (float*)d_out, cnt);
}
